// ChebNet_14439680049854
// MI455X (gfx1250) — matrix operand order repaired and checked by reading, not compiled, not run on hardware
//
#include <hip/hip_runtime.h>
#include <hip/hip_bf16.h>

// ---------------------------------------------------------------------------
// ChebNet on MI455X (gfx1250)
//   - ELL (degree<=8) gather Laplacian, combine fused, zero atomics
//   - single fused WMMA GEMM per layer: K = 6*F_in, bf16 convert in-register,
//     B fragments staged in LDS then bulk-preloaded to registers so the
//     8 WMMAs issue back-to-back, C written once (bias-initialized in regs)
// ---------------------------------------------------------------------------

typedef __attribute__((ext_vector_type(16))) __bf16 v16bf;
typedef __attribute__((ext_vector_type(8)))  float  v8f;

__device__ __forceinline__ __bf16 f2bf(float f) {
  __hip_bfloat16 h = __float2bfloat16(f);
  __bf16 r;
  __builtin_memcpy(&r, &h, sizeof(r));
  return r;
}

// ---------------- ELL build (once per graph level) -------------------------

// append each edge into its destination's padded (8-wide) adjacency row
__global__ void k_ell_build(const int* __restrict__ src, const int* __restrict__ dst,
                            const float* __restrict__ ea, int E,
                            int* __restrict__ cursor,
                            int* __restrict__ srcPad, float* __restrict__ eaPad) {
  int e = blockIdx.x * blockDim.x + threadIdx.x;
  if (e >= E) return;
  int d = dst[e];
  int p = atomicAdd(&cursor[d], 1);
  if (p < 8) {
    srcPad[(long long)d * 8 + p] = src[e];
    eaPad[(long long)d * 8 + p]  = ea[e];
  }
}

// ---------------- Laplacian apply + fused Chebyshev combine ----------------
// out[n][f] = acc            (sub == nullptr)   where acc = sum_d ea*v[src][f]
// out[n][f] = 2*acc - sub[n][f]                 (otherwise)
__global__ void k_lap_ell(const float* __restrict__ v,
                          const int* __restrict__ srcPad,
                          const float* __restrict__ eaPad,
                          const float* __restrict__ sub,
                          float* __restrict__ out,
                          long long total, int F) {
  long long idx = (long long)blockIdx.x * blockDim.x + threadIdx.x;
  if (idx >= total) return;
  long long n = idx / F;
  int f = (int)(idx % F);
  const int*   sp = srcPad + n * 8;
  const float* ep = eaPad + n * 8;
  float acc = 0.0f;
#pragma unroll
  for (int d = 0; d < 8; ++d)
    acc += ep[d] * v[(long long)sp[d] * F + f];
  out[idx] = sub ? (2.0f * acc - sub[idx]) : acc;
}

// x [N][3] -> xpad [N][32] fp32 (zero padded)
__global__ void k_pad32(const float* __restrict__ in, float* __restrict__ out,
                        long long total) {
  long long i = (long long)blockIdx.x * blockDim.x + threadIdx.x;
  if (i >= total) return;
  long long n = i >> 5;
  int c = (int)(i & 31);
  out[i] = (c < 3) ? in[n * 3 + c] : 0.0f;
}

// Pack W [6][Fin][Fout] fp32 into per-lane WMMA B fragments (bf16):
// frag[k][kk][tile][lane][j]; lane<16: col=lane+tile*16, kin=kk*32+(j&7)+(j>>3)*16
//                             lane>=16: col=lane-16+tile*16, kin=kk*32+8+(j&7)+(j>>3)*16
__global__ void k_pack_wfrag(const float* __restrict__ W,
                             __hip_bfloat16* __restrict__ frag,
                             int Fin, int Fout, int numKK, int numTiles, int total) {
  int idx = blockIdx.x * blockDim.x + threadIdx.x;
  if (idx >= total) return;
  int j = idx & 15;
  int lane = (idx >> 4) & 31;
  int rest = idx >> 9;
  int tile = rest % numTiles;
  int rest2 = rest / numTiles;
  int kk = rest2 % numKK;
  int k  = rest2 / numKK;
  int col = tile * 16 + (lane & 15);
  int kin = kk * 32 + (lane >> 4) * 8 + (j & 7) + (j >> 3) * 16;
  float v = (kin < Fin && col < Fout) ? W[((long long)k * Fin + kin) * Fout + col] : 0.0f;
  frag[idx] = __float2bfloat16(v);
}

// ---------------- fused WMMA GEMM ------------------------------------------
// C[N][ldc] = bias + sum_{k=0..5} T_k[N][Wd] @ W_k    (T_0 = A0, T_k = Tp[k-1])
template <int NT>
__global__ void __launch_bounds__(128)
k_wmma_gemm(const float* __restrict__ A0, const float* __restrict__ Tp,
            long long planeStride, const __hip_bfloat16* __restrict__ Bfrag,
            const float* __restrict__ bias, float* __restrict__ C,
            int N, int Wd, int ldc, int Fout) {
  const int numKK = Wd >> 5;                  // Wd = 32 or 128
  const int fragK = numKK * NT * 512;         // bf16 elems per k
  __shared__ __align__(32) __bf16 sB[4 * NT * 512];

  const int lane = threadIdx.x;               // 0..31
  const int rowTile = blockIdx.x * blockDim.y + threadIdx.y;
  const int m  = lane & 15;
  const int hi = lane >> 4;
  const int tid = threadIdx.y * 32 + lane;

  v8f c[NT];
#pragma unroll
  for (int t = 0; t < NT; ++t) {
    int col = t * 16 + m;
    float bv = (col < Fout) ? bias[col] : 0.0f;
#pragma unroll
    for (int r = 0; r < 8; ++r) c[t][r] = bv;
  }

  const bool active = (rowTile * 16 < N);
  const long long rowOff = (long long)(rowTile * 16 + m) * Wd;

#pragma unroll 1
  for (int k = 0; k < 6; ++k) {
    // cooperative stage of this k's B fragments into LDS (all 128 threads)
    {
      const uint4* src = reinterpret_cast<const uint4*>(Bfrag + (long long)k * fragK);
      uint4* dst = reinterpret_cast<uint4*>(sB);
      const int nvec = fragK / 8;              // 8 bf16 per uint4
      for (int i = tid; i < nvec; i += 128) dst[i] = src[i];
    }
    __syncthreads();
    if (active) {
      const float* Abase = (k == 0) ? A0 : (Tp + (long long)(k - 1) * planeStride);
      const float* arow = Abase + rowOff;
#pragma unroll 1
      for (int kk = 0; kk < numKK; ++kk) {
        v8f alo = *reinterpret_cast<const v8f*>(arow + kk * 32 + hi * 8);
        v8f ahi = *reinterpret_cast<const v8f*>(arow + kk * 32 + 16 + hi * 8);
        __builtin_prefetch(arow + (kk + 1) * 32, 0, 1);   // global_prefetch_b8
        v16bf a;
#pragma unroll
        for (int j = 0; j < 8; ++j) {
          a[j]     = f2bf(alo[j]);
          a[j + 8] = f2bf(ahi[j]);
        }
        const __bf16* sbase = sB + (long long)kk * NT * 512 + lane * 16;
        // bulk-preload all NT B fragments -> one dscnt wait, then a clean
        // back-to-back WMMA burst
        v16bf b[NT];
#pragma unroll
        for (int t = 0; t < NT; ++t)
          b[t] = *reinterpret_cast<const v16bf*>(sbase + t * 512);
#pragma unroll
        for (int t = 0; t < NT; ++t)
          c[t] = __builtin_amdgcn_wmma_f32_16x16x32_bf16(
              false, a, false, b[t], (short)0, c[t], false, false);
      }
    }
    __syncthreads();
  }

  if (!active) return;
#pragma unroll
  for (int t = 0; t < NT; ++t) {
#pragma unroll
    for (int r = 0; r < 8; ++r)
      C[(long long)(rowTile * 16 + hi * 8 + r) * ldc + t * 16 + m] = c[t][r];
  }
}

// ---------------- BatchNorm / ReLU / pooling / softmax ---------------------

__global__ void k_bn_stats(const float* __restrict__ x, float* __restrict__ stats,
                           int N, int F) {
  int f = threadIdx.x;          // launched with F threads
  float s = 0.0f, sq = 0.0f;
  for (int n = blockIdx.x; n < N; n += gridDim.x) {
    float v = x[(long long)n * F + f];
    s += v; sq += v * v;
  }
  atomicAdd(&stats[f], s);
  atomicAdd(&stats[F + f], sq);
}

__global__ void k_bn_relu(float* __restrict__ x, const float* __restrict__ stats,
                          const float* __restrict__ g, const float* __restrict__ bt,
                          long long total, int N, int F) {
  long long i = (long long)blockIdx.x * blockDim.x + threadIdx.x;
  if (i >= total) return;
  int f = (int)(i % F);
  float invN = 1.0f / (float)N;
  float mu = stats[f] * invN;
  float var = stats[F + f] * invN - mu * mu;
  float y = g[f] * (x[i] - mu) * rsqrtf(var + 1e-5f) + bt[f];
  x[i] = fmaxf(y, 0.0f);
}

__global__ void k_relu(float* __restrict__ x, long long n) {
  long long i = (long long)blockIdx.x * blockDim.x + threadIdx.x;
  if (i < n) x[i] = fmaxf(x[i], 0.0f);
}

// segment max via int-bit atomicMax (valid: post-ReLU values >= 0, dst zeroed)
__global__ void k_seg_max(const float* __restrict__ x, const int* __restrict__ cl,
                          float* __restrict__ out, long long total, int F) {
  long long i = (long long)blockIdx.x * blockDim.x + threadIdx.x;
  if (i >= total) return;
  int n = (int)(i / F);
  int f = (int)(i % F);
  atomicMax(reinterpret_cast<int*>(&out[(long long)cl[n] * F + f]),
            __float_as_int(x[i]));
}

__global__ void k_logsoftmax(const float* __restrict__ g, float* __restrict__ out) {
  int b = threadIdx.x;
  if (b >= 8) return;
  float mx = -3.4e38f;
  for (int c = 0; c < 10; ++c) mx = fmaxf(mx, g[b * 16 + c]);
  float sum = 0.0f;
  for (int c = 0; c < 10; ++c) sum += __expf(g[b * 16 + c] - mx);
  float ls = __logf(sum);
  for (int c = 0; c < 10; ++c) out[b * 10 + c] = g[b * 16 + c] - mx - ls;
}

// ---------------- host orchestration ---------------------------------------

static inline long long cdivll(long long a, long long b) { return (a + b - 1) / b; }

// Build padded adjacency for one graph level.
static void build_ell(const int* ei, const float* ea, int E, int N,
                      int* cursor, int* srcPad, float* eaPad, hipStream_t s) {
  hipMemsetAsync(cursor, 0, (size_t)N * sizeof(int), s);
  hipMemsetAsync(srcPad, 0, (size_t)N * 8 * sizeof(int), s);
  hipMemsetAsync(eaPad, 0, (size_t)N * 8 * sizeof(float), s);
  k_ell_build<<<(unsigned)cdivll(E, 256), 256, 0, s>>>(ei, ei + E, ea, E,
                                                       cursor, srcPad, eaPad);
}

// act: fp32 [N][Wd] (padded for layer 1). planes: 5 contiguous [N][Wd] planes.
static void cheb_layer(const float* act, int Wd,
                       const int* srcPad, const float* eaPad,
                       const float* W, const float* bias, int Fin, int Fout,
                       float* out, int ldc,
                       float* planes, __hip_bfloat16* wfrag,
                       int N, hipStream_t s) {
  const int numKK = Wd / 32;
  const int numTiles = (Fout + 15) / 16;      // 8 or 1
  const int fragK = numKK * numTiles * 512;
  const int fragTotal = 6 * fragK;

  k_pack_wfrag<<<(unsigned)cdivll(fragTotal, 256), 256, 0, s>>>(
      W, wfrag, Fin, Fout, numKK, numTiles, fragTotal);

  const long long plane = (long long)N * Wd;
  auto lapf = [&](const float* src, float* dst, const float* sub) {
    k_lap_ell<<<(unsigned)cdivll(plane, 256), 256, 0, s>>>(
        src, srcPad, eaPad, sub, dst, plane, Wd);
  };
  float* P0 = planes;
  float* P1 = planes + plane;
  float* P2 = planes + 2 * plane;
  float* P3 = planes + 3 * plane;
  float* P4 = planes + 4 * plane;

  lapf(act, P0, nullptr);   // T1 = L T0
  lapf(P0, P1, act);        // T2 = 2 L T1 - T0
  lapf(P1, P2, P0);         // T3
  lapf(P2, P3, P1);         // T4
  lapf(P3, P4, P2);         // T5

  dim3 blk(32, 4);
  dim3 grd((unsigned)cdivll(N / 16, 4), 1);
  if (numTiles == 8)
    k_wmma_gemm<8><<<grd, blk, 0, s>>>(act, planes, plane, wfrag, bias, out,
                                       N, Wd, ldc, Fout);
  else
    k_wmma_gemm<1><<<grd, blk, 0, s>>>(act, planes, plane, wfrag, bias, out,
                                       N, Wd, ldc, Fout);
}

static void bn_relu_host(float* x, const float* g, const float* bt, float* stats,
                         int N, int F, hipStream_t s) {
  hipMemsetAsync(stats, 0, 2 * F * sizeof(float), s);
  k_bn_stats<<<512, F, 0, s>>>(x, stats, N, F);
  long long tot = (long long)N * F;
  k_bn_relu<<<(unsigned)cdivll(tot, 256), 256, 0, s>>>(x, stats, g, bt, tot, N, F);
}

static void pool_max_host(const float* x, const int* cl, float* out,
                          int Nin, int Nout, int F, hipStream_t s) {
  hipMemsetAsync(out, 0, (size_t)Nout * F * sizeof(float), s);
  long long tot = (long long)Nin * F;
  k_seg_max<<<(unsigned)cdivll(tot, 256), 256, 0, s>>>(x, cl, out, tot, F);
}

extern "C" void kernel_launch(void* const* d_in, const int* in_sizes, int n_in,
                              void* d_out, int out_size, void* d_ws, size_t ws_size,
                              hipStream_t stream) {
  (void)n_in; (void)out_size; (void)ws_size;

  const float* x   = (const float*)d_in[0];
  const int*   ei1 = (const int*)d_in[1];  const float* ea1 = (const float*)d_in[2];
  const int*   ei2 = (const int*)d_in[3];  const float* ea2 = (const float*)d_in[4];
  const int*   ei3 = (const int*)d_in[5];  const float* ea3 = (const float*)d_in[6];
  const int*   cl1 = (const int*)d_in[7];
  const int*   cl2 = (const int*)d_in[8];
  const int*   cl3 = (const int*)d_in[9];
  const int*   cl4 = (const int*)d_in[10];
  const int*   bfi = (const int*)d_in[11];
  const float* W[6]; const float* bb[6];
  for (int i = 0; i < 6; ++i) { W[i] = (const float*)d_in[12 + 2 * i];
                                bb[i] = (const float*)d_in[13 + 2 * i]; }
  const float* gma[5]; const float* bta[5];
  for (int i = 0; i < 5; ++i) { gma[i] = (const float*)d_in[24 + 2 * i];
                                bta[i] = (const float*)d_in[25 + 2 * i]; }

  const int N1 = in_sizes[7];          // 196608
  const int N2 = in_sizes[8];          // 49152
  const int N3 = in_sizes[9];          // 12288
  const int N4 = in_sizes[10];         // 3072
  const int N5 = in_sizes[11];         // 512
  const int E1 = in_sizes[1] / 2, E2 = in_sizes[3] / 2, E3 = in_sizes[5] / 2;

  // workspace bump allocator (256B aligned)
  char* ws = (char*)d_ws;
  size_t off = 0;
  auto alloc = [&](size_t bytes) -> void* {
    off = (off + 255) & ~(size_t)255;
    void* p = ws + off;
    off += bytes;
    return p;
  };
  const size_t P = (size_t)N1 * 128;
  float* f0     = (float*)alloc(P * sizeof(float));
  float* f1     = (float*)alloc(P * sizeof(float));
  float* planes = (float*)alloc(5 * P * sizeof(float));
  float* xpad   = (float*)alloc((size_t)N1 * 32 * sizeof(float));
  __hip_bfloat16* wfrag =
      (__hip_bfloat16*)alloc((size_t)6 * 4 * 8 * 32 * 16 * sizeof(__hip_bfloat16));
  int*   cursor = (int*)alloc((size_t)N1 * sizeof(int));
  int*   e1s = (int*)alloc((size_t)N1 * 8 * sizeof(int));
  float* e1a = (float*)alloc((size_t)N1 * 8 * sizeof(float));
  int*   e2s = (int*)alloc((size_t)N2 * 8 * sizeof(int));
  float* e2a = (float*)alloc((size_t)N2 * 8 * sizeof(float));
  int*   e3s = (int*)alloc((size_t)N3 * 8 * sizeof(int));
  float* e3a = (float*)alloc((size_t)N3 * 8 * sizeof(float));
  float* stats = (float*)alloc(256 * sizeof(float));
  float* gbuf  = (float*)alloc(128 * sizeof(float));

  hipStream_t s = stream;

  // padded adjacency per level (built once, reused by all laps of that level)
  build_ell(ei1, ea1, E1, N1, cursor, e1s, e1a, s);
  build_ell(ei2, ea2, E2, N2, cursor, e2s, e2a, s);
  build_ell(ei3, ea3, E3, N3, cursor, e3s, e3a, s);

  // pad x [N1][3] -> [N1][32] so layer 1 runs the uniform path
  {
    long long tot = (long long)N1 * 32;
    k_pad32<<<(unsigned)cdivll(tot, 256), 256, 0, s>>>(x, xpad, tot);
  }

  // ----- layer 1: [N1,3(pad32)] -> [N1,128]
  cheb_layer(xpad, 32, e1s, e1a, W[0], bb[0], 3, 128, f1, 128, planes, wfrag, N1, s);
  bn_relu_host(f1, gma[0], bta[0], stats, N1, 128, s);
  // ----- layer 2
  cheb_layer(f1, 128, e1s, e1a, W[1], bb[1], 128, 128, f0, 128, planes, wfrag, N1, s);
  bn_relu_host(f0, gma[1], bta[1], stats, N1, 128, s);
  pool_max_host(f0, cl1, f1, N1, N2, 128, s);
  // ----- layer 3
  cheb_layer(f1, 128, e2s, e2a, W[2], bb[2], 128, 128, f0, 128, planes, wfrag, N2, s);
  bn_relu_host(f0, gma[2], bta[2], stats, N2, 128, s);
  // ----- layer 4
  cheb_layer(f0, 128, e2s, e2a, W[3], bb[3], 128, 128, f1, 128, planes, wfrag, N2, s);
  bn_relu_host(f1, gma[3], bta[3], stats, N2, 128, s);
  pool_max_host(f1, cl2, f0, N2, N3, 128, s);
  // ----- layer 5
  cheb_layer(f0, 128, e3s, e3a, W[4], bb[4], 128, 128, f1, 128, planes, wfrag, N3, s);
  bn_relu_host(f1, gma[4], bta[4], stats, N3, 128, s);
  // ----- layer 6: 128 -> 10 (ldc=16 padded), ReLU only
  cheb_layer(f1, 128, e3s, e3a, W[5], bb[5], 128, 10, f0, 16, planes, wfrag, N3, s);
  {
    long long tot = (long long)N3 * 16;
    k_relu<<<(unsigned)cdivll(tot, 256), 256, 0, s>>>(f0, tot);
  }
  // ----- pools: spatial, orientation, global
  pool_max_host(f0, cl3, f1, N3, N4, 16, s);
  pool_max_host(f1, cl4, f0, N4, N5, 16, s);
  pool_max_host(f0, bfi, gbuf, N5, 8, 16, s);
  // ----- log_softmax -> d_out [8][10]
  k_logsoftmax<<<1, 32, 0, s>>>(gbuf, (float*)d_out);
}
